// FeedForward_43396349559335
// MI455X (gfx1250) — matrix operand order repaired and checked by reading, not compiled, not run on hardware
//
#include <hip/hip_runtime.h>

typedef _Float16 v16h __attribute__((ext_vector_type(16)));
typedef _Float16 v8h  __attribute__((ext_vector_type(8)));
typedef float    v8f  __attribute__((ext_vector_type(8)));
typedef int      v4i  __attribute__((ext_vector_type(4)));

#define T_TOK 64
#define GS 128
#define LDS_STRIDE 136   // 128 + 8 halves pad -> 68-dword row stride, conflict-free b128 reads

union V16 { v16h v; v8h h[2]; _Float16 a[16]; };

#if defined(__AMDGCN__) && __has_builtin(__builtin_amdgcn_global_load_async_to_lds_b128) && __has_builtin(__builtin_amdgcn_s_wait_asynccnt)
#define USE_ASYNC_LDS 1
typedef __attribute__((address_space(1))) v4i* gas_v4i_ptr;   // global
typedef __attribute__((address_space(3))) v4i* las_v4i_ptr;   // LDS
#else
#define USE_ASYNC_LDS 0
#endif

__global__ void k_zero(float* __restrict__ p, int n) {
  int i = blockIdx.x * blockDim.x + threadIdx.x;
  if (i < n) p[i] = 0.0f;
}

__global__ void k_cvt_f32_f16(const float* __restrict__ x, _Float16* __restrict__ xh, int n) {
  int i = blockIdx.x * blockDim.x + threadIdx.x;
  if (i < n) xh[i] = (_Float16)x[i];
}

__global__ void k_silu_mul(const float* __restrict__ g, const float* __restrict__ u,
                           _Float16* __restrict__ o, int n) {
  int i = blockIdx.x * blockDim.x + threadIdx.x;
  if (i < n) {
    float gv = g[i];
    float sg = gv / (1.0f + __expf(-gv));
    o[i] = (_Float16)(sg * u[i]);
  }
}

// Cooperative fill of one 64x128 f16 A tile (group `grp`) into LDS.
// 1024 16-byte chunks / 256 threads = 4 chunks each; 16 consecutive threads
// cover one row's 256 bytes -> fully coalesced.
__device__ __forceinline__ void fill_a_tile(_Float16* __restrict__ dstLds,
                                            const _Float16* __restrict__ A,
                                            int K, int grp, int tid) {
  #pragma unroll
  for (int r = 0; r < 4; ++r) {
    const int cid = tid + r * 256;          // chunk id 0..1023
    const int row = cid >> 4;               // 0..63
    const int c8  = cid & 15;               // 16-byte chunk within row
    const _Float16* gsrc = A + (size_t)row * K + (size_t)grp * GS + c8 * 8;
    _Float16* l = dstLds + row * LDS_STRIDE + c8 * 8;
#if USE_ASYNC_LDS
    __builtin_amdgcn_global_load_async_to_lds_b128(
        (gas_v4i_ptr)(void*)gsrc, (las_v4i_ptr)(void*)l, 0, 0);
#else
    *(v8h*)l = *(const v8h*)gsrc;
#endif
  }
}

__device__ __forceinline__ void lds_fill_fence() {
#if USE_ASYNC_LDS
  __builtin_amdgcn_s_wait_asynccnt(0);
#endif
  __syncthreads();
}

// Generic group-quantized GEMM:  dst[64, N] += A[64, K] @ dequant(q[N, K]).T
// Block = 8 waves sharing an LDS-staged A tile; wave owns a 16-wide N tile and
// all 64 M rows (4 WMMA fragments). grid.y slices K (groupsPerSlice 128-groups).
__global__ __launch_bounds__(256) void k_gemm_q4(
    const _Float16* __restrict__ A,   // [64, K] f16 row-major
    const int*      __restrict__ q,   // [N, K] int32 (values 0..15)
    const float*    __restrict__ s,   // [N, K/128]
    const int*      __restrict__ zp,  // [N, K/128]
    float*          __restrict__ dst, // [64, N] f32, atomic accumulation
    int K, int N, int groupsPerSlice)
{
  __shared__ _Float16 smem[2][64 * LDS_STRIDE];   // double-buffered A tile (34 KB)

  const int gH    = K / GS;
  const int tid   = threadIdx.x;
  const int wave  = tid >> 5;
  const int lid   = tid & 31;
  const int lrow  = lid & 15;        // B column / C column within tile
  const int lhalf = lid >> 4;        // K-half selector
  const int n0    = (blockIdx.x * 8 + wave) * 16;
  const int myN   = n0 + lrow;       // weight row this lane dequantizes
  const int g0    = blockIdx.y * groupsPerSlice;

  const int* qrow = q + (size_t)myN * K;

  v8f acc[4] = {};

  fill_a_tile(smem[0], A, K, g0, tid);
  lds_fill_fence();

  for (int gi = 0; gi < groupsPerSlice; ++gi) {
    const int grp = g0 + gi;
    if (gi + 1 < groupsPerSlice)                       // async prefetch next tile
      fill_a_tile(smem[(gi + 1) & 1], A, K, grp + 1, tid);

    const float sv = s[(size_t)myN * gH + grp];
    const float zv = (float)zp[(size_t)myN * gH + grp];
    const _Float16* sm = smem[gi & 1];

    // streamed, non-temporal weight loads; pipeline one 32-K block ahead
    v4i qc[4], qn[4];
    {
      const v4i* qp = (const v4i*)(qrow + grp * GS + lhalf * 16);
      qc[0] = __builtin_nontemporal_load(qp + 0);
      qc[1] = __builtin_nontemporal_load(qp + 1);
      qc[2] = __builtin_nontemporal_load(qp + 2);
      qc[3] = __builtin_nontemporal_load(qp + 3);
    }

    #pragma unroll
    for (int kk = 0; kk < 4; ++kk) {
      if (kk < 3) {
        const v4i* qp = (const v4i*)(qrow + grp * GS + (kk + 1) * 32 + lhalf * 16);
        qn[0] = __builtin_nontemporal_load(qp + 0);
        qn[1] = __builtin_nontemporal_load(qp + 1);
        qn[2] = __builtin_nontemporal_load(qp + 2);
        qn[3] = __builtin_nontemporal_load(qp + 3);
      }

      int qv[16];
      ((v4i*)qv)[0] = qc[0];
      ((v4i*)qv)[1] = qc[1];
      ((v4i*)qv)[2] = qc[2];
      ((v4i*)qv)[3] = qc[3];

      V16 bf;
      #pragma unroll
      for (int j = 0; j < 16; ++j)
        bf.a[j] = (_Float16)(((float)qv[j] - zv) * sv);

      // 4 M-tiles reuse the dequantized B fragment; A fragments from LDS
      #pragma unroll
      for (int m = 0; m < 4; ++m) {
        const _Float16* ar = sm + (16 * m + lrow) * LDS_STRIDE + kk * 32 + lhalf * 16;
        V16 af;
        af.h[0] = *(const v8h*)ar;          // K = kk*32 + lhalf*16 + [0..7]
        af.h[1] = *(const v8h*)(ar + 8);    // K = kk*32 + lhalf*16 + 8 + [0..7]
        acc[m] = __builtin_amdgcn_wmma_f32_16x16x32_f16(
            false, af.v, false, bf.v, (short)0, acc[m], false, false);
      }

      qc[0] = qn[0]; qc[1] = qn[1]; qc[2] = qn[2]; qc[3] = qn[3];
    }

    lds_fill_fence();   // next buffer ready + everyone done with current
  }

  // C layout: VGPR r -> (M = 16m + r + lhalf*8, Ncol = n0 + lrow)
  #pragma unroll
  for (int m = 0; m < 4; ++m) {
    #pragma unroll
    for (int r = 0; r < 8; ++r) {
      const int row = 16 * m + r + lhalf * 8;
      __hip_atomic_fetch_add(&dst[(size_t)row * N + n0 + lrow], acc[m][r],
                             __ATOMIC_RELAXED, __HIP_MEMORY_SCOPE_AGENT);
    }
  }
}

extern "C" void kernel_launch(void* const* d_in, const int* in_sizes, int n_in,
                              void* d_out, int out_size, void* d_ws, size_t ws_size,
                              hipStream_t stream) {
  const float* x   = (const float*)d_in[0];
  const int*   q1  = (const int*)  d_in[1];
  const float* s1  = (const float*)d_in[2];
  const int*   zp1 = (const int*)  d_in[3];
  const int*   q3  = (const int*)  d_in[4];
  const float* s3  = (const float*)d_in[5];
  const int*   zp3 = (const int*)  d_in[6];
  const int*   q2  = (const int*)  d_in[7];
  const float* s2  = (const float*)d_in[8];
  const int*   zp2 = (const int*)  d_in[9];
  float* out = (float*)d_out;

  const int H = 4096, I = 11008, T = T_TOK;

  char* ws = (char*)d_ws;
  _Float16* xh   = (_Float16*)ws;                               // 64*4096  f16  (512 KB)
  float*    gbuf = (float*)(ws + 524288);                       // 64*11008 f32  (2.69 MB)
  float*    ubuf = (float*)(ws + 524288 + 2818048);             // 64*11008 f32
  _Float16* x13  = (_Float16*)(ws + 524288 + 2 * 2818048);      // 64*11008 f16  (1.34 MB)

  // zero the fp32 accumulators (gbuf and ubuf are contiguous) and the output
  k_zero<<<(2 * T * I + 255) / 256, 256, 0, stream>>>(gbuf, 2 * T * I);
  k_zero<<<(T * H + 255) / 256, 256, 0, stream>>>(out, T * H);

  // x -> f16 once; all GEMM A-reads come from LDS staged out of this buffer
  k_cvt_f32_f16<<<(T * H + 255) / 256, 256, 0, stream>>>(x, xh, T * H);

  // gate and up projections: K=4096 (32 groups), 8 K-slices x 4 groups
  k_gemm_q4<<<dim3(I / 128, 8), 256, 0, stream>>>(xh, q1, s1, zp1, gbuf, H, I, 4);
  k_gemm_q4<<<dim3(I / 128, 8), 256, 0, stream>>>(xh, q3, s3, zp3, ubuf, H, I, 4);

  // fused silu(gate) * up -> f16 activation for the down projection
  k_silu_mul<<<(T * I + 255) / 256, 256, 0, stream>>>(gbuf, ubuf, x13, T * I);

  // down projection: K=11008 (86 groups), 43 K-slices x 2 groups
  k_gemm_q4<<<dim3(H / 128, 43), 256, 0, stream>>>(x13, q2, s2, zp2, out, I, H, 2);
}
